// SeparateMappingModule_81853486728035
// MI455X (gfx1250) — matrix operand order repaired and checked by reading, not compiled, not run on hardware
//
#include <hip/hip_runtime.h>
#include <math.h>

typedef __attribute__((ext_vector_type(16))) _Float16 v16h;
typedef __attribute__((ext_vector_type(8)))  float    v8f;

#define NROWS 50000
#define KDIM  2048
#define ODIM  512
#define MTILES 3125            // NROWS / 16 (exact)
#define KTILES 64              // KDIM / 32
#define CHUNKS 125
#define ROWS_PER_CHUNK 400     // 125 * 400 = 50000

// ---------------- Pass 1: per-column online softmax stats (single pass over X) ----------
__global__ __launch_bounds__(256) void col_stats_partial(
    const float* __restrict__ X, float* __restrict__ pm,
    float* __restrict__ ps, float* __restrict__ pt) {
  int col   = blockIdx.x * 256 + threadIdx.x;      // 8 blocks * 256 = 2048 cols
  int chunk = blockIdx.y;
  const float* p = X + (size_t)chunk * ROWS_PER_CHUNK * KDIM + col;
  float m = -INFINITY, s = 0.0f, t = 0.0f;
  for (int r = 0; r < ROWS_PER_CHUNK; ++r) {
    float x = p[(size_t)r * KDIM];                 // coalesced across threads
    if (x > m) { float c = expf(m - x); s *= c; t *= c; m = x; }
    float e = expf(x - m);
    s += e; t += x * e;
  }
  int o = chunk * KDIM + col;
  pm[o] = m; ps[o] = s; pt[o] = t;
}

__global__ __launch_bounds__(256) void entropy_from_partials(
    const float* __restrict__ pm, const float* __restrict__ ps,
    const float* __restrict__ pt, float* __restrict__ ent) {
  int col = blockIdx.x * 256 + threadIdx.x;
  float M = -INFINITY;
  for (int c = 0; c < CHUNKS; ++c) M = fmaxf(M, pm[c * KDIM + col]);
  float S = 0.0f, T = 0.0f;
  for (int c = 0; c < CHUNKS; ++c) {
    float sc = expf(pm[c * KDIM + col] - M);
    S += ps[c * KDIM + col] * sc;
    T += pt[c * KDIM + col] * sc;
  }
  // entropy(nats) = lse - E[x] = M + log(S) - T/S  (monotone in entropy-bits)
  ent[col] = M + logf(S) - T / S;
}

// ---------------- Sort: stable argsort(-entropy) via one-block bitonic -------------------
__global__ __launch_bounds__(1024) void bitonic_sort_perm(
    const float* __restrict__ ent, int* __restrict__ perm) {
  __shared__ float key[KDIM];
  __shared__ int   val[KDIM];
  for (int i = threadIdx.x; i < KDIM; i += 1024) { key[i] = -ent[i]; val[i] = i; }
  __syncthreads();
  for (int k = 2; k <= KDIM; k <<= 1) {
    for (int j = k >> 1; j > 0; j >>= 1) {
      for (int i = threadIdx.x; i < KDIM; i += 1024) {
        int l = i ^ j;
        if (l > i) {
          float ki = key[i], kl = key[l];
          int   vi = val[i], vl = val[l];
          bool asc = ((i & k) == 0);
          bool gt  = (ki > kl) || (ki == kl && vi > vl);   // lexicographic (key, idx)
          if (gt == asc) { key[i] = kl; key[l] = ki; val[i] = vl; val[l] = vi; }
        }
      }
      __syncthreads();
    }
  }
  for (int i = threadIdx.x; i < KDIM; i += 1024) perm[i] = val[i];
}

// ---------------- Pack W_inf||W_red -> f16, WMMA B-fragment layout -----------------------
// Fragment id t = ((kt*32 + ng)*32 + lane); lane: n = ng*16 + (lane&15), K = kt*32 + (lane>>4)*16 + j
__global__ __launch_bounds__(256) void pack_B(
    const float* __restrict__ Winf, const float* __restrict__ Wred,
    _Float16* __restrict__ Wp) {
  int t    = blockIdx.x * 256 + threadIdx.x;       // 65536 fragment rows
  int lane = t & 31;
  int n    = ((t >> 5) & 31) * 16 + (lane & 15);
  int kb   = (t >> 10) * 32 + (lane >> 4) * 16;
  v16h frag;
#pragma unroll
  for (int j = 0; j < 16; ++j) {
    int k = kb + j;
    float w = (k < ODIM) ? Winf[(size_t)k * ODIM + n]
                         : Wred[(size_t)(k - ODIM) * ODIM + n];
    frag[j] = (_Float16)w;
  }
  *(v16h*)(Wp + (size_t)t * 16) = frag;
}

// ---------------- Gather+convert X -> f16, WMMA A-fragment layout ------------------------
// Fragment id t = ((mt*64 + kt)*32 + lane); lane: M = mt*16 + (lane&15),
// K offset = (lane>>4)*16 + j ; column = perm[kt*32 + Koff]
__global__ __launch_bounds__(256) void pack_A(
    const float* __restrict__ X, const int* __restrict__ perm,
    _Float16* __restrict__ Xp) {
  size_t t = (size_t)blockIdx.x * 256 + threadIdx.x;   // 6,400,000 fragment rows
  int lane = (int)(t & 31);
  int kt   = (int)((t >> 5) & 63);
  int mt   = (int)(t >> 11);
  int row  = mt * 16 + (lane & 15);
  int hi8  = (lane >> 4) * 16;
  const float* xr = X + (size_t)row * KDIM;
  int kb = kt * 32;
  v16h frag;
#pragma unroll
  for (int j = 0; j < 16; ++j) {
    int koff = hi8 + j;
    int col  = perm[kb + koff];
    frag[j]  = (_Float16)xr[col];
  }
  *(v16h*)(Xp + t * 16) = frag;
}

// ---------------- Fused dual-GEMM + tanh with v_wmma_f32_16x16x32_f16 --------------------
// Grid: 3125 WGs (one per 16-row M-tile), 8 waves/WG, wave w owns output cols [w*64, w*64+64)
__global__ __launch_bounds__(256) void gemm_wmma(
    const _Float16* __restrict__ Xp, const _Float16* __restrict__ Wp,
    const float* __restrict__ binf, const float* __restrict__ bred,
    float* __restrict__ out) {
  int tid  = threadIdx.x;
  int lane = tid & 31;
  int w    = tid >> 5;                 // 0..7 : 64-col group
  int mt   = blockIdx.x;               // 0..3124
  int n0   = w * 64;

  const v16h* Ap = (const v16h*)Xp + (size_t)mt * (KTILES * 32) + lane;
  const v16h* Bp = (const v16h*)Wp + (size_t)(w * 4) * 32 + lane;

  v8f acc[4], res[4];
#pragma unroll
  for (int s = 0; s < 4; ++s)
#pragma unroll
    for (int v = 0; v < 8; ++v) acc[s][v] = 0.0f;

  // K tiles 0..15 : info block (K = 0..511)
  for (int kt = 0; kt < 16; ++kt) {
    v16h a = Ap[kt * 32];
#pragma unroll
    for (int s = 0; s < 4; ++s) {
      v16h b = Bp[kt * 1024 + s * 32];
      acc[s] = __builtin_amdgcn_wmma_f32_16x16x32_f16(
          false, a, false, b, (short)0, acc[s], false, false);
    }
  }
  int ncol = n0 + (lane & 15);
#pragma unroll
  for (int s = 0; s < 4; ++s) {
    float bi = binf[ncol + s * 16];
#pragma unroll
    for (int v = 0; v < 8; ++v) { res[s][v] = tanhf(acc[s][v] + bi); acc[s][v] = 0.0f; }
  }

  // K tiles 16..63 : red block (K = 512..2047)
  for (int kt = 16; kt < KTILES; ++kt) {
    v16h a = Ap[kt * 32];
#pragma unroll
    for (int s = 0; s < 4; ++s) {
      v16h b = Bp[kt * 1024 + s * 32];
      acc[s] = __builtin_amdgcn_wmma_f32_16x16x32_f16(
          false, a, false, b, (short)0, acc[s], false, false);
    }
  }

  int rbase = mt * 16 + (lane >> 4) * 8;   // C/D layout: vgpr v -> row v (+8 for high lanes)
#pragma unroll
  for (int s = 0; s < 4; ++s) {
    float br = bred[ncol + s * 16];
    int col  = ncol + s * 16;
#pragma unroll
    for (int v = 0; v < 8; ++v) {
      float r = res[s][v] + tanhf(acc[s][v] + br);
      out[(size_t)(rbase + v) * ODIM + col] = r;
    }
  }
}

// ---------------- Host launcher ----------------------------------------------------------
extern "C" void kernel_launch(void* const* d_in, const int* in_sizes, int n_in,
                              void* d_out, int out_size, void* d_ws, size_t ws_size,
                              hipStream_t stream) {
  const float* X    = (const float*)d_in[0];   // (50000, 2048)
  const float* Winf = (const float*)d_in[1];   // (512, 512)
  const float* binf = (const float*)d_in[2];   // (512,)
  const float* Wred = (const float*)d_in[3];   // (1536, 512)
  const float* bred = (const float*)d_in[4];   // (512,)
  float* out = (float*)d_out;                  // (50000, 512)

  char* ws = (char*)d_ws;
  size_t off = 0;
  _Float16* Xp = (_Float16*)(ws + off); off += (size_t)MTILES * KTILES * 512 * sizeof(_Float16); // 204.8 MB
  _Float16* Wp = (_Float16*)(ws + off); off += (size_t)KDIM * ODIM * sizeof(_Float16);           // 2 MB
  int*   perm = (int*)  (ws + off); off += KDIM * sizeof(int);
  float* ent  = (float*)(ws + off); off += KDIM * sizeof(float);
  float* pm   = (float*)(ws + off); off += (size_t)CHUNKS * KDIM * sizeof(float);
  float* ps   = (float*)(ws + off); off += (size_t)CHUNKS * KDIM * sizeof(float);
  float* pt   = (float*)(ws + off); off += (size_t)CHUNKS * KDIM * sizeof(float);
  (void)ws_size; (void)in_sizes; (void)n_in; (void)out_size;

  col_stats_partial<<<dim3(KDIM / 256, CHUNKS), 256, 0, stream>>>(X, pm, ps, pt);
  entropy_from_partials<<<KDIM / 256, 256, 0, stream>>>(pm, ps, pt, ent);
  bitonic_sort_perm<<<1, 1024, 0, stream>>>(ent, perm);
  pack_B<<<(KTILES * 32 * 32) / 256, 256, 0, stream>>>(Winf, Wred, Wp);
  pack_A<<<(MTILES * KTILES * 32) / 256, 256, 0, stream>>>(X, perm, Xp);
  gemm_wmma<<<MTILES, 256, 0, stream>>>(Xp, Wp, binf, bred, out);
}
